// AttentionDynamicModel_82257213653619
// MI455X (gfx1250) — matrix operand order repaired and checked by reading, not compiled, not run on hardware
//
#include <hip/hip_runtime.h>

typedef __attribute__((ext_vector_type(16))) _Float16 v16h;
typedef __attribute__((ext_vector_type(8)))  _Float16 v8h;
typedef __attribute__((ext_vector_type(8)))  float    v8f;

#define DIM     128
#define NHEAD   8
#define NSC     1008            // max score rows (63 tiles of 16)
#define NPAD    1024            // K extent for Ebar WMMA (32 chunks of 32)
#define SESTR   136             // sE row stride (halves): 16B-aligned, bank-friendly
#define WSTR    1032            // weight row stride (halves)
#define NEG_INF (-1e9f)
#define MASKED_SCORE (-60000.0f)   // f16-representable "-inf"; keeps all-masked case stable

// ---------------------------------------------------------------------------
// Kernel 1: per-batch precompute.
//   Q[b,:]    = context[b]@Wq_ctx + bq + step[b]@Wq_step
//   qk[b,d,h] = sum_{jj<16} Wk[d, h*16+jj] * Q[b, h*16+jj]   (f16, cols 8..15 = 0)
// Stored [B][128][16] f16 — directly consumable as a WMMA B matrix.
// ---------------------------------------------------------------------------
__global__ __launch_bounds__(128) void qk_precompute(
    const float* __restrict__ context, const float* __restrict__ step_ctx,
    const float* __restrict__ Wq_ctx,  const float* __restrict__ bq,
    const float* __restrict__ Wq_step, const float* __restrict__ Wk,
    _Float16* __restrict__ qk)
{
  const int b = blockIdx.x;
  const int t = threadIdx.x;
  __shared__ float sC[DIM], sS[DIM], sQ[DIM];
  sC[t] = context[(size_t)b*DIM + t];
  sS[t] = step_ctx[(size_t)b*DIM + t];
  __syncthreads();

  float q = bq[t];
  #pragma unroll 4
  for (int d = 0; d < DIM; ++d)
    q += sC[d]*Wq_ctx[d*DIM + t] + sS[d]*Wq_step[d*DIM + t];
  sQ[t] = q;
  __syncthreads();

  const int d = t;
  _Float16* o = qk + ((size_t)b*DIM + d)*16;
  #pragma unroll
  for (int h = 0; h < NHEAD; ++h) {
    float acc = 0.f;
    #pragma unroll
    for (int jj = 0; jj < 16; ++jj)
      acc += Wk[d*DIM + h*16 + jj] * sQ[h*16 + jj];
    o[h] = (_Float16)acc;
  }
  #pragma unroll
  for (int h = NHEAD; h < 16; ++h) o[h] = (_Float16)0.f;
}

// ---------------------------------------------------------------------------
// Kernel 2: one workgroup per batch. Streams E[b] from HBM exactly once
// (f32 -> f16 into LDS), computing per-head scores with WMMA on the fly.
// Then: per-head softmax (f16 weights, transposed), attention-weighted
// embedding sum (WMMA), ctx/mha/m2 GEMVs, compat GEMV (WMMA), final softmax.
// ---------------------------------------------------------------------------
__global__ __launch_bounds__(256) void fused_attn(
    const float* __restrict__ emb,  const int* __restrict__ mask,
    const float* __restrict__ Wv,   const float* __restrict__ Wout,
    const float* __restrict__ Wk_tanh, const _Float16* __restrict__ qk,
    float* __restrict__ out, int N)
{
  __shared__ __align__(16) _Float16 sE[NPAD * SESTR];     // 278528 B
  __shared__ __align__(16) _Float16 sScoreH[NSC * NHEAD]; //  16128 B
  __shared__ __align__(16) _Float16 sW[NHEAD * WSTR];     //  16512 B (reused as float compat[])
  __shared__ __align__(16) _Float16 sQk[DIM * 16];        //   4096 B
  __shared__ float sEbar[NHEAD * DIM];                    //   4096 B
  __shared__ float sCtx[DIM], sMha[DIM], sM2[DIM];
  __shared__ float sHsum[NHEAD];
  __shared__ float sRed[12];

  const int b    = blockIdx.x;
  const int t    = threadIdx.x;
  const int wave = t >> 5;
  const int lane = t & 31;

  const int ntiles = (N + 15) >> 4;
  const int nsc    = ntiles * 16;

  // ---- zero E pad rows [nsc, NPAD) so 0-weight WMMA terms can't see NaNs ---
  for (int i = t; i < (NPAD - nsc) * DIM; i += 256) {
    const int r = nsc + (i >> 7);
    const int c = i & 127;
    sE[(size_t)r*SESTR + c] = (_Float16)0.f;
  }

  // ---- stage qk[b]; build wave-uniform WMMA B fragments (cols = heads) ----
  for (int i = t; i < DIM*16; i += 256) sQk[i] = qk[(size_t)b*DIM*16 + i];
  __syncthreads();

  v16h bfrag[4];
  {
    const int c    = lane & 15;
    const int koff = (lane < 16) ? 0 : 8;        // B 32x16: lanes 0-15 K 0..7,16..23
    #pragma unroll
    for (int kc = 0; kc < 4; ++kc)
      #pragma unroll
      for (int i = 0; i < 16; ++i)
        bfrag[kc][i] = sQk[(kc*32 + koff + i + (i & 8))*16 + c];
  }

  // ---- stream E once, scores via WMMA -------------------------------------
  // A 16x32 f16: lane L(0-15)=row L, halves {k..k+7, k+16..k+23}; lane L+16
  // = row L, halves {k+8..k+15, k+24..k+31}.
  const int row = lane & 15;
  const int grp = (lane < 16) ? 0 : 1;

  for (int tt = wave; tt < ntiles; tt += 8) {
    const int n0 = tt * 16;
    const int n  = n0 + row;
    const bool valid = (n < N);
    const int  nclamp = valid ? n : (N - 1);
    const float* rp = emb + ((size_t)b*N + nclamp)*DIM;

    v8f acc = {0.f,0.f,0.f,0.f,0.f,0.f,0.f,0.f};
    #pragma unroll
    for (int kc = 0; kc < 4; ++kc) {
      const int base = kc*32 + grp*8;
      float4 a0 = *reinterpret_cast<const float4*>(rp + base);
      float4 a1 = *reinterpret_cast<const float4*>(rp + base + 4);
      float4 c0 = *reinterpret_cast<const float4*>(rp + base + 16);
      float4 c1 = *reinterpret_cast<const float4*>(rp + base + 20);
      v16h a;
      a[0]=(_Float16)a0.x; a[1]=(_Float16)a0.y; a[2]=(_Float16)a0.z; a[3]=(_Float16)a0.w;
      a[4]=(_Float16)a1.x; a[5]=(_Float16)a1.y; a[6]=(_Float16)a1.z; a[7]=(_Float16)a1.w;
      a[8]=(_Float16)c0.x; a[9]=(_Float16)c0.y; a[10]=(_Float16)c0.z; a[11]=(_Float16)c0.w;
      a[12]=(_Float16)c1.x; a[13]=(_Float16)c1.y; a[14]=(_Float16)c1.z; a[15]=(_Float16)c1.w;
      if (!valid) {
        #pragma unroll
        for (int i = 0; i < 16; ++i) a[i] = (_Float16)0.f;
      }
      v8h lo, hi;
      #pragma unroll
      for (int i = 0; i < 8; ++i) { lo[i] = a[i]; hi[i] = a[i+8]; }
      *reinterpret_cast<__shared__ v8h*>(&sE[(size_t)n*SESTR + base])      = lo;
      *reinterpret_cast<__shared__ v8h*>(&sE[(size_t)n*SESTR + base + 16]) = hi;
      // EXEC all-1s here (no divergent branch around WMMA)
      acc = __builtin_amdgcn_wmma_f32_16x16x32_f16(
              false, a, false, bfrag[kc], (short)0, acc, false, false);
    }

    // D: lane holds col h=lane%16, rows v + (lane<16?0:8)
    const int h = lane & 15;
    if (h < NHEAD) {
      const int rbase = (lane < 16) ? 0 : 8;
      #pragma unroll
      for (int v = 0; v < 8; ++v) {
        const int nn = n0 + rbase + v;
        float sval;
        if (nn >= N) sval = -__builtin_inff();
        else {
          const int mk = mask[(size_t)b*N + nn];
          sval = (mk == 1) ? MASKED_SCORE : acc[v] * 0.25f;  // 1/sqrt(DH)
        }
        sScoreH[nn*NHEAD + h] = (_Float16)sval;
      }
    }
  }
  __syncthreads();

  // ---- per-head softmax (wave h owns head h); unnormalized f16 weights ----
  {
    const int h = wave;
    float m = -__builtin_inff();
    for (int n = lane; n < nsc; n += 32)
      m = fmaxf(m, (float)sScoreH[n*NHEAD + h]);
    #pragma unroll
    for (int off = 16; off > 0; off >>= 1) m = fmaxf(m, __shfl_xor(m, off, 32));
    float sum = 0.f;
    for (int n = lane; n < nsc; n += 32) {
      float p = __expf((float)sScoreH[n*NHEAD + h] - m);
      sW[h*WSTR + n] = (_Float16)p;
      sum += p;
    }
    #pragma unroll
    for (int off = 16; off > 0; off >>= 1) sum += __shfl_xor(sum, off, 32);
    if (lane == 0) sHsum[h] = sum;
    for (int n = nsc + lane; n < WSTR; n += 32)   // zero K-padding
      sW[h*WSTR + n] = (_Float16)0.f;
  }
  __syncthreads();

  // ---- Ebar via WMMA: Ebar(16h x 128d) = W(16 x 1024n) @ E(1024n x 128d) ---
  // Wave w owns d-columns [16w, 16w+16); accumulates 32 K-chunks.
  {
    const int c0    = wave * 16;
    const int lrow  = lane & 15;
    const int akoff = (lane < 16) ? 0 : 8;    // A-layout K offset
    const int bkoff = (lane < 16) ? 0 : 8;    // B-layout K offset
    v8f acc = {0.f,0.f,0.f,0.f,0.f,0.f,0.f,0.f};
    for (int ch = 0; ch < NPAD/32; ++ch) {
      const int k0 = ch * 32;
      v8h wlo = *reinterpret_cast<__shared__ v8h*>(&sW[lrow*WSTR + k0 + akoff]);
      v8h whi = *reinterpret_cast<__shared__ v8h*>(&sW[lrow*WSTR + k0 + akoff + 16]);
      v16h a;
      #pragma unroll
      for (int i = 0; i < 8; ++i) { a[i] = wlo[i]; a[8+i] = whi[i]; }
      v16h bm;
      #pragma unroll
      for (int i = 0; i < 16; ++i)
        bm[i] = sE[(size_t)(k0 + bkoff + i + (i & 8))*SESTR + c0 + lrow];
      acc = __builtin_amdgcn_wmma_f32_16x16x32_f16(
              false, a, false, bm, (short)0, acc, false, false);
    }
    if (lane < 16) {                      // rows 0..7 = heads (8..15 are zero)
      const int d = c0 + lrow;
      #pragma unroll
      for (int v = 0; v < 8; ++v)
        sEbar[v*DIM + d] = acc[v] / sHsum[v];
    }
  }
  __syncthreads();

  // ---- ctx = Ebar @ Wv (head-blocked), mha = ctx @ Wout, m2 = Wk_tanh@mha --
  if (t < DIM) {
    const int j = t, h = j >> 4;
    float s = 0.f;
    #pragma unroll 4
    for (int d = 0; d < DIM; ++d) s += sEbar[h*DIM + d] * Wv[d*DIM + j];
    sCtx[j] = s;
  }
  __syncthreads();
  if (t < DIM) {
    const int j = t;
    float s = 0.f;
    #pragma unroll 4
    for (int d = 0; d < DIM; ++d) s += sCtx[d] * Wout[d*DIM + j];
    sMha[j] = s;
  }
  __syncthreads();
  if (t < DIM) {
    const int d = t;
    float s = 0.f;
    #pragma unroll 4
    for (int j = 0; j < DIM; ++j) s += Wk_tanh[d*DIM + j] * sMha[j];
    sM2[d] = s;
  }
  __syncthreads();

  // ---- compat[n] = E[n,:]·m2 via WMMA (m2 in B column 0) ------------------
  float* sCompat = reinterpret_cast<float*>(sW);   // sW fully consumed by Ebar
  {
    v16h bm2[4];
    const int c    = lane & 15;
    const int koff = (lane < 16) ? 0 : 8;
    #pragma unroll
    for (int kc = 0; kc < 4; ++kc)
      #pragma unroll
      for (int i = 0; i < 16; ++i)
        bm2[kc][i] = (c == 0) ? (_Float16)sM2[kc*32 + koff + i + (i & 8)] : (_Float16)0.f;

    const int grpoff = (lane < 16) ? 0 : 8;
    for (int tt = wave; tt < ntiles; tt += 8) {
      const int n0 = tt*16;
      const int n  = n0 + (lane & 15);
      v8f acc = {0.f,0.f,0.f,0.f,0.f,0.f,0.f,0.f};
      #pragma unroll
      for (int kc = 0; kc < 4; ++kc) {
        const int base = kc*32 + grpoff;
        v8h lo = *reinterpret_cast<__shared__ v8h*>(&sE[(size_t)n*SESTR + base]);
        v8h hi = *reinterpret_cast<__shared__ v8h*>(&sE[(size_t)n*SESTR + base + 16]);
        v16h a;
        #pragma unroll
        for (int i = 0; i < 8; ++i) { a[i] = lo[i]; a[8+i] = hi[i]; }
        acc = __builtin_amdgcn_wmma_f32_16x16x32_f16(
                false, a, false, bm2[kc], (short)0, acc, false, false);
      }
      if ((lane & 15) == 0) {             // col 0 holds the dot products
        const int rbase = (lane < 16) ? 0 : 8;
        #pragma unroll
        for (int v = 0; v < 8; ++v)
          sCompat[n0 + rbase + v] = acc[v];
      }
    }
  }
  __syncthreads();

  // ---- tanh clip + mask + stable softmax over n ---------------------------
  float lmax = -__builtin_inff();
  for (int n = t; n < N; n += 256) {
    const int mk = mask[(size_t)b*N + n];
    float cv = (mk == 1) ? NEG_INF
                         : tanhf(sCompat[n] * 0.08838834764831845f) * 10.0f;
    sCompat[n] = cv;
    lmax = fmaxf(lmax, cv);
  }
  #pragma unroll
  for (int off = 16; off > 0; off >>= 1) lmax = fmaxf(lmax, __shfl_xor(lmax, off, 32));
  if (lane == 0) sRed[wave] = lmax;
  __syncthreads();
  if (t == 0) {
    float mm = -__builtin_inff();
    #pragma unroll
    for (int w = 0; w < 8; ++w) mm = fmaxf(mm, sRed[w]);
    sRed[8] = mm;
  }
  __syncthreads();
  const float cmax = sRed[8];

  float lsum = 0.f;
  for (int n = t; n < N; n += 256) {
    float e = __expf(sCompat[n] - cmax);
    sCompat[n] = e;
    lsum += e;
  }
  #pragma unroll
  for (int off = 16; off > 0; off >>= 1) lsum += __shfl_xor(lsum, off, 32);
  if (lane == 0) sRed[wave] = lsum;
  __syncthreads();
  if (t == 0) {
    float ss = 0.f;
    #pragma unroll
    for (int w = 0; w < 8; ++w) ss += sRed[w];
    sRed[9] = ss;
  }
  __syncthreads();
  const float inv = 1.0f / sRed[9];
  for (int n = t; n < N; n += 256)
    out[(size_t)b*N + n] = sCompat[n] * inv;
}

// ---------------------------------------------------------------------------
extern "C" void kernel_launch(void* const* d_in, const int* in_sizes, int n_in,
                              void* d_out, int out_size, void* d_ws, size_t ws_size,
                              hipStream_t stream) {
  const float* emb   = (const float*)d_in[0];
  const float* ctx   = (const float*)d_in[1];
  const float* stepc = (const float*)d_in[2];
  const float* Wk    = (const float*)d_in[3];
  const float* Wkt   = (const float*)d_in[4];
  const float* Wv    = (const float*)d_in[5];
  const float* Wqc   = (const float*)d_in[6];
  const float* bqc   = (const float*)d_in[7];
  const float* Wqs   = (const float*)d_in[8];
  const float* Wo    = (const float*)d_in[9];
  const int*   mask  = (const int*)d_in[10];
  float* out = (float*)d_out;

  const int B = in_sizes[1] / DIM;                 // context is [B, D]
  int N = in_sizes[0] / (B * DIM);                 // embeddings [B, N, D]
  if (N > NSC) N = NSC;

  _Float16* qk = (_Float16*)d_ws;                  // [B][128][16] f16 = 2 MB

  qk_precompute<<<B, DIM, 0, stream>>>(ctx, stepc, Wqc, bqc, Wqs, Wk, qk);
  fused_attn<<<B, 256, 0, stream>>>(emb, mask, Wv, Wo, Wkt, qk, out, N);
}
